// TriangleAttention_15616501088402
// MI455X (gfx1250) — matrix operand order repaired and checked by reading, not compiled, not run on hardware
//
#include <hip/hip_runtime.h>
#include <hip/hip_bf16.h>
#include <cstdint>

#define B_SZ  4
#define N_SZ  128
#define D_SZ  256
#define H_SZ  8
#define DK_SZ 32

typedef __bf16 bf16_t;
typedef __attribute__((ext_vector_type(16))) __bf16 v16bf;
typedef __attribute__((ext_vector_type(8)))  __bf16 v8bf;
typedef __attribute__((ext_vector_type(8)))  float  v8f;
typedef uint32_t su32x4 __attribute__((ext_vector_type(4)));
typedef uint32_t su32x8 __attribute__((ext_vector_type(8)));

__device__ inline v8f v8f_zero() {
  v8f z;
#pragma unroll
  for (int i = 0; i < 8; ++i) z[i] = 0.0f;
  return z;
}

// --- Tensor Data Mover: 2D tile load global->LDS (D# per ISA 08 §8.3/8.4) ----
// tile: tile1 rows of tile0 elements (bf16, data_size=2B), global row stride
// `stride0` (elements). Destination LDS is packed [tile1][tile0].
__device__ inline void tdm_load_2d_bf16(uint32_t lds_off, const void* gptr,
                                        uint32_t tdim0, uint32_t tdim1,
                                        uint32_t tile0, uint32_t tile1,
                                        uint32_t stride0) {
  const uint64_t ga = (uint64_t)(uintptr_t)gptr;
  su32x4 g0;
  g0[0] = 1u;                                        // count=1 (valid user D#)
  g0[1] = lds_off;                                   // lds_addr (bytes)
  g0[2] = (uint32_t)(ga & 0xFFFFFFFFu);              // global_addr[31:0]
  g0[3] = (uint32_t)((ga >> 32) & 0x1FFFFFFu)        // global_addr[56:32]
          | (2u << 30);                              // type=2 ("image")
  su32x8 g1;
  g1[0] = (1u << 16);                                // data_size=1 -> 2 bytes
  g1[1] = (tdim0 & 0xFFFFu) << 16;                   // tensor_dim0[15:0]
  g1[2] = (tdim0 >> 16) | ((tdim1 & 0xFFFFu) << 16); // dim0[31:16], dim1[15:0]
  g1[3] = (tdim1 >> 16) | ((tile0 & 0xFFFFu) << 16); // dim1[31:16], tile_dim0
  g1[4] = (tile1 & 0xFFFFu);                         // tile_dim1 (tile_dim2=0)
  g1[5] = stride0;                                   // tensor_dim0_stride[31:0]
  g1[6] = 0u;                                        // stride0[47:32], stride1 lo
  g1[7] = 0u;
  asm volatile("tensor_load_to_lds %0, %1" :: "s"(g0), "s"(g1) : "memory");
}

// --- WMMA operand loaders (wave32, v_wmma_f32_16x16x32_bf16 layouts) ---------
// A operand 16x32 (MxK), source row-major [M][K] with leading dim lda.
__device__ inline v16bf load_A(const bf16_t* base, int lda, int lane) {
  const int m  = lane & 15;
  const int hf = lane >> 4;
  const bf16_t* row = base + (size_t)m * lda + 8 * hf;
  v8bf lo = *(const v8bf*)(row);       // K = 8*hf + 0..7
  v8bf hi = *(const v8bf*)(row + 16);  // K = 16 + 8*hf + 0..7
  v16bf a;
#pragma unroll
  for (int i = 0; i < 8; ++i) { a[i] = lo[i]; a[8 + i] = hi[i]; }
  return a;
}

// B operand 32x16 (KxN), source stored as [N][K] row-major, K = 8*hf+0..7, 16+8*hf+0..7.
__device__ inline v16bf load_B(const bf16_t* base, int ldb, int lane) {
  const int n  = lane & 15;
  const int hf = lane >> 4;
  const bf16_t* col = base + (size_t)n * ldb + 8 * hf;
  v8bf lo = *(const v8bf*)(col);
  v8bf hi = *(const v8bf*)(col + 16);
  v16bf b;
#pragma unroll
  for (int i = 0; i < 8; ++i) { b[i] = lo[i]; b[8 + i] = hi[i]; }
  return b;
}

// --- elementwise converts ----------------------------------------------------
__global__ void cvt_x_kernel(const float* __restrict__ x, bf16_t* __restrict__ xb, int n4) {
  int t = blockIdx.x * blockDim.x + threadIdx.x;
  if (t < n4) {
    float4 v = ((const float4*)x)[t];
    bf16_t* o = xb + (size_t)t * 4;
    o[0] = (bf16_t)v.x; o[1] = (bf16_t)v.y; o[2] = (bf16_t)v.z; o[3] = (bf16_t)v.w;
  }
}

// W [256][Nw] f32 -> Wt [Nw][256] bf16
__global__ void cvt_wt_kernel(const float* __restrict__ W, bf16_t* __restrict__ Wt, int Nw) {
  int k = threadIdx.x;
  int n = blockIdx.x;
  Wt[(size_t)n * 256 + k] = (bf16_t)W[(size_t)k * Nw + n];
}

// --- triangle bias: fbias[b,h,j,k] = bias[b,0,h,j,k] + sum_d x[b,j,k,d]*Wproj[d,h]
__global__ __launch_bounds__(256) void tri_bias_kernel(
    const float* __restrict__ x, const float* __restrict__ bias,
    const float* __restrict__ Wproj, float* __restrict__ fbias) {
  const int lane = threadIdx.x & 31;
  const int wave = threadIdx.x >> 5;
  const int pos  = blockIdx.x * 8 + wave;           // < B*N*N = 65536
  const float* xr = x + (size_t)pos * D_SZ;
  const int h = lane & 7;
  const int g = lane >> 3;
  float acc = 0.0f;
#pragma unroll 8
  for (int t = 0; t < 64; ++t) {
    int d = t * 4 + g;
    acc += xr[d] * Wproj[d * H_SZ + h];
  }
  acc += __shfl_xor(acc, 8, 32);
  acc += __shfl_xor(acc, 16, 32);
  if (lane < 8) {
    int b = pos >> 14, j = (pos >> 7) & 127, kk = pos & 127;
    size_t bi = (((size_t)b * H_SZ + lane) * N_SZ + j) * N_SZ + kk;
    fbias[bi] = bias[bi] + acc;
  }
}

// --- GEMM: C[M][256] = A[M][256] @ Bt[256][256]^T + bias, TDM-staged weights -
template <bool BF16OUT>
__global__ __launch_bounds__(256) void gemm_kernel(
    const bf16_t* __restrict__ A, const bf16_t* __restrict__ Bt,
    const float* __restrict__ bias, void* __restrict__ Cout,
    int M, int Nn, int Kk) {
  __shared__ bf16_t Wlds[128 * 256];   // 64 KB: Bt slice [n0..n0+128)[0..256)
  const int lane = threadIdx.x & 31;
  const int wave = threadIdx.x >> 5;
  const int m0 = blockIdx.x * 128 + wave * 16;
  const int n0 = blockIdx.y * 128;

  if (wave == 0) {
    // one DMA: 128*256 bf16 (contiguous) -> LDS, then drain TENSORcnt
    tdm_load_2d_bf16((uint32_t)(uintptr_t)&Wlds[0],
                     Bt + (size_t)n0 * 256,
                     /*tdim0=*/32768u, /*tdim1=*/1u,
                     /*tile0=*/32768u, /*tile1=*/1u,
                     /*stride0=*/32768u);
    __builtin_amdgcn_s_wait_tensorcnt(0);
  }
  __syncthreads();

  v8f acc[8];
#pragma unroll
  for (int t = 0; t < 8; ++t) acc[t] = v8f_zero();

  for (int kc = 0; kc < Kk; kc += 32) {
    v16bf a = load_A(A + (size_t)m0 * Kk + kc, Kk, lane);
#pragma unroll
    for (int t = 0; t < 8; ++t) {
      v16bf b = load_B(&Wlds[(size_t)(t * 16) * 256 + kc], 256, lane);
      acc[t] = __builtin_amdgcn_wmma_f32_16x16x32_bf16(
          false, a, false, b, (short)0, acc[t], false, false);
    }
  }

  const int hf = lane >> 4, nn = lane & 15;
#pragma unroll
  for (int t = 0; t < 8; ++t) {
    const int ncol = n0 + t * 16 + nn;
    const float bv = bias[ncol];
#pragma unroll
    for (int r = 0; r < 8; ++r) {
      const int row = m0 + r + 8 * hf;
      const float vv = acc[t][r] + bv;
      if (BF16OUT) ((bf16_t*)Cout)[(size_t)row * Nn + ncol] = (bf16_t)vv;
      else         ((float*)Cout)[(size_t)row * Nn + ncol]  = vv;
    }
  }
}

// --- fused attention per (b,i,h): S=QK^T*s+bias, softmax, O=PV --------------
__global__ __launch_bounds__(256) void attn_kernel(
    const bf16_t* __restrict__ qp, const bf16_t* __restrict__ kp,
    const bf16_t* __restrict__ vp, const float* __restrict__ fbias,
    bf16_t* __restrict__ op) {
  __shared__ bf16_t Plds[N_SZ * N_SZ];   // 32 KB: P in [j][k]
  __shared__ bf16_t Vl[N_SZ * DK_SZ];    //  8 KB: V tile [k][d] (TDM dest)
  __shared__ bf16_t Vt[DK_SZ * N_SZ];    //  8 KB: V^T [d][k]

  const int idx = blockIdx.x;            // (b*N + i)*H + h
  const int h  = idx % H_SZ;
  const int bi = idx / H_SZ;
  const int i  = bi % N_SZ;
  const int b  = bi / N_SZ;
  const size_t rowbase = ((size_t)(b * N_SZ + i)) * N_SZ;

  const int lane = threadIdx.x & 31;
  const int wave = threadIdx.x >> 5;

  if (wave == 0) {
    // 2D TDM: 128 rows (k) x 32 elems (d), global row stride 256 elements
    tdm_load_2d_bf16((uint32_t)(uintptr_t)&Vl[0],
                     vp + rowbase * D_SZ + h * DK_SZ,
                     /*tdim0=*/(uint32_t)DK_SZ, /*tdim1=*/(uint32_t)N_SZ,
                     /*tile0=*/(uint32_t)DK_SZ, /*tile1=*/(uint32_t)N_SZ,
                     /*stride0=*/(uint32_t)D_SZ);
    __builtin_amdgcn_s_wait_tensorcnt(0);
  }
  __syncthreads();
  // transpose V tile in LDS: Vt[d][k] = Vl[k][d]
  for (int t = threadIdx.x; t < DK_SZ * N_SZ; t += blockDim.x) {
    int d = t >> 7, kk = t & 127;
    Vt[d * N_SZ + kk] = Vl[kk * DK_SZ + d];
  }
  __syncthreads();

  const int jb = wave * 16;
  const int hf = lane >> 4, nn = lane & 15;

  // scores: one WMMA per 16x16 tile (WMMA K=32 == DK)
  v16bf aq = load_A(qp + (rowbase + jb) * D_SZ + h * DK_SZ, D_SZ, lane);
  v8f sc[8];
#pragma unroll
  for (int t = 0; t < 8; ++t) {
    v16bf bk = load_B(kp + (rowbase + t * 16) * D_SZ + h * DK_SZ, D_SZ, lane);
    sc[t] = __builtin_amdgcn_wmma_f32_16x16x32_bf16(
        false, aq, false, bk, (short)0, v8f_zero(), false, false);
  }

  const float scale = 0.17677669529663687f;  // 1/sqrt(32)
  const float* fb = fbias + ((size_t)b * H_SZ + h) * N_SZ * N_SZ;

#pragma unroll
  for (int r = 0; r < 8; ++r) {
    const int j = jb + r + 8 * hf;
    const float* fbr = fb + (size_t)j * N_SZ;
    float mx = -3.0e38f;
#pragma unroll
    for (int t = 0; t < 8; ++t) {
      float s = sc[t][r] * scale + fbr[t * 16 + nn];
      sc[t][r] = s;
      mx = fmaxf(mx, s);
    }
#pragma unroll
    for (int msk = 1; msk < 16; msk <<= 1) mx = fmaxf(mx, __shfl_xor(mx, msk, 32));
    float sm = 0.0f;
#pragma unroll
    for (int t = 0; t < 8; ++t) {
      float e = __expf(sc[t][r] - mx);
      sc[t][r] = e;
      sm += e;
    }
#pragma unroll
    for (int msk = 1; msk < 16; msk <<= 1) sm += __shfl_xor(sm, msk, 32);
    const float inv = 1.0f / sm;
#pragma unroll
    for (int t = 0; t < 8; ++t)
      Plds[(size_t)j * N_SZ + t * 16 + nn] = (bf16_t)(sc[t][r] * inv);
  }
  __syncthreads();

  // O = P @ V : 16 rows x 32 cols per wave = 2 tiles x 4 K-chunks
#pragma unroll
  for (int dblk = 0; dblk < 2; ++dblk) {
    v8f acc = v8f_zero();
#pragma unroll
    for (int kb = 0; kb < 4; ++kb) {
      v16bf ap = load_A(&Plds[(size_t)jb * N_SZ + kb * 32], N_SZ, lane);
      v16bf bv = load_B(&Vt[(size_t)(dblk * 16) * N_SZ + kb * 32], N_SZ, lane);
      acc = __builtin_amdgcn_wmma_f32_16x16x32_bf16(
          false, ap, false, bv, (short)0, acc, false, false);
    }
#pragma unroll
    for (int r = 0; r < 8; ++r) {
      const int j = jb + r + 8 * hf;
      const int d = dblk * 16 + nn;
      op[(rowbase + j) * D_SZ + h * DK_SZ + d] = (bf16_t)acc[r];
    }
  }
}

extern "C" void kernel_launch(void* const* d_in, const int* in_sizes, int n_in,
                              void* d_out, int out_size, void* d_ws, size_t ws_size,
                              hipStream_t stream) {
  const float* x     = (const float*)d_in[0];
  const float* bias  = (const float*)d_in[1];
  const float* Wproj = (const float*)d_in[2];
  const float* Wq    = (const float*)d_in[3];
  const float* bq    = (const float*)d_in[4];
  const float* Wk    = (const float*)d_in[5];
  const float* bk    = (const float*)d_in[6];
  const float* Wv    = (const float*)d_in[7];
  const float* bv    = (const float*)d_in[8];
  const float* Wo    = (const float*)d_in[9];
  const float* bo    = (const float*)d_in[10];

  const size_t XE = (size_t)B_SZ * N_SZ * N_SZ * D_SZ;  // 16,777,216
  const size_t WE = 256 * 256;
  const size_t FB = (size_t)B_SZ * H_SZ * N_SZ * N_SZ;  // 524,288

  char* ws = (char*)d_ws;
  bf16_t* xb  = (bf16_t*)ws;              ws += XE * 2;
  bf16_t* wtq = (bf16_t*)ws;              ws += WE * 2;
  bf16_t* wtk = (bf16_t*)ws;              ws += WE * 2;
  bf16_t* wtv = (bf16_t*)ws;              ws += WE * 2;
  bf16_t* wto = (bf16_t*)ws;              ws += WE * 2;
  bf16_t* qb  = (bf16_t*)ws;              ws += XE * 2;
  bf16_t* kb_ = (bf16_t*)ws;              ws += XE * 2;
  bf16_t* vb  = (bf16_t*)ws;              ws += XE * 2;
  bf16_t* ob  = (bf16_t*)ws;              ws += XE * 2;
  float*  fbias = (float*)ws;             ws += FB * 4;

  // 1) converts
  {
    int n4 = (int)(XE / 4);
    cvt_x_kernel<<<dim3((n4 + 255) / 256), dim3(256), 0, stream>>>(x, xb, n4);
    cvt_wt_kernel<<<dim3(256), dim3(256), 0, stream>>>(Wq, wtq, 256);
    cvt_wt_kernel<<<dim3(256), dim3(256), 0, stream>>>(Wk, wtk, 256);
    cvt_wt_kernel<<<dim3(256), dim3(256), 0, stream>>>(Wv, wtv, 256);
    cvt_wt_kernel<<<dim3(256), dim3(256), 0, stream>>>(Wo, wto, 256);
  }
  // 2) triangle bias (f32 path)
  tri_bias_kernel<<<dim3(65536 / 8), dim3(256), 0, stream>>>(x, bias, Wproj, fbias);

  // 3) QKV projections: [65536 x 256] @ [256 x 256], bf16 out
  {
    dim3 g(65536 / 128, 256 / 128), blk(256);
    gemm_kernel<true><<<g, blk, 0, stream>>>(xb, wtq, bq, qb,  65536, 256, 256);
    gemm_kernel<true><<<g, blk, 0, stream>>>(xb, wtk, bk, kb_, 65536, 256, 256);
    gemm_kernel<true><<<g, blk, 0, stream>>>(xb, wtv, bv, vb,  65536, 256, 256);
  }

  // 4) attention: one block per (b, i, h)
  attn_kernel<<<dim3(B_SZ * N_SZ * H_SZ), dim3(256), 0, stream>>>(qb, kb_, vb, fbias, ob);

  // 5) output projection, f32 out to d_out
  {
    dim3 g(65536 / 128, 256 / 128), blk(256);
    gemm_kernel<false><<<g, blk, 0, stream>>>(ob, wto, bo, d_out, 65536, 256, 256);
  }
}
